// GDLAttention_41583873360559
// MI455X (gfx1250) — matrix operand order repaired and checked by reading, not compiled, not run on hardware
//
#include <hip/hip_runtime.h>
#include <math.h>

// ---------------------------------------------------------------------------
// Types
// ---------------------------------------------------------------------------
typedef __attribute__((ext_vector_type(16))) __bf16        v16bf;
typedef __attribute__((ext_vector_type(8)))  float         v8f;
typedef __attribute__((ext_vector_type(4)))  unsigned int  u32x4;

#define D_MODEL 1024
#define N_HEADS 16
#define DK      64
#define HIDDEN  512
#define BATCH   8
#define SEQ     1024
#define NROWS   (BATCH * SEQ)   // 8192

// ---------------------------------------------------------------------------
// bf16 helpers (round-to-nearest-even)
// ---------------------------------------------------------------------------
__device__ inline __bf16 f2bf(float f) {
    union { float f; unsigned u; } x; x.f = f;
    unsigned r = x.u + 0x7FFFu + ((x.u >> 16) & 1u);
    union { unsigned short s; __bf16 b; } y; y.s = (unsigned short)(r >> 16);
    return y.b;
}
__device__ inline float bf2f(__bf16 b) {
    union { unsigned short s; __bf16 b; } y; y.b = b;
    union { unsigned u; float f; } x; x.u = ((unsigned)y.s) << 16;
    return x.f;
}

// ---------------------------------------------------------------------------
// Async global -> LDS copy (CDNA5 ASYNCcnt path).
// LDS aperture keeps the byte offset in addr[31:0] (ISA 10.2), so the
// truncated generic pointer is a valid DS address for the async dest VGPR.
// ---------------------------------------------------------------------------
__device__ inline void async_copy_b128(void* lds_dst, const void* gsrc) {
    unsigned da = (unsigned)(uintptr_t)lds_dst;
    asm volatile("global_load_async_to_lds_b128 %0, %1, off"
                 :: "v"(da), "v"(gsrc) : "memory");
}
__device__ inline void wait_async0() {
    asm volatile("s_wait_asynccnt 0x0" ::: "memory");
}

// ---------------------------------------------------------------------------
// WMMA wrapper + fragment loaders (ISA 7.12.2 layouts, wave32)
// ---------------------------------------------------------------------------
__device__ inline v8f wmma_bf16(v16bf a, v16bf b, v8f c) {
    return __builtin_amdgcn_wmma_f32_16x16x32_bf16(
        /*neg_a=*/false, a, /*neg_b=*/false, b,
        /*c_mod=*/(short)0, c, /*reuse_a=*/false, /*reuse_b=*/false);
}

// A fragment 16x32 bf16: lane = row (lane&15), lane>>4 selects K-halves.
// VGPR0-3 hold K = g*8 + 0..7, VGPR4-7 hold K = 16 + g*8 + 0..7.
__device__ inline v16bf load_frag_a(const __bf16* lds, int row_base, int k_base, int ldk) {
    int lane = threadIdx.x & 31;
    int r = lane & 15, g = lane >> 4;
    const __bf16* p = lds + (size_t)(row_base + r) * ldk + k_base + g * 8;
    union { u32x4 u[2]; v16bf v; } t;
    t.u[0] = *(const u32x4*)(p);        // K g*8 + 0..7
    t.u[1] = *(const u32x4*)(p + 16);   // K 16 + g*8 + 0..7
    return t.v;
}

// B fragment 32x16 (KxN) bf16: lane = col (lane&15), K = g*8 + 0..7 then 16 + g*8 + 0..7.
__device__ inline v16bf load_frag_b(const __bf16* lds, int n_base, int k_base, int ldk) {
    int lane = threadIdx.x & 31;
    int n = lane & 15, g = lane >> 4;
    const __bf16* p = lds + (size_t)(n_base + n) * ldk + k_base + g * 8;
    union { u32x4 u[2]; v16bf v; } t;
    t.u[0] = *(const u32x4*)(p);
    t.u[1] = *(const u32x4*)(p + 16);
    return t.v;
}

// 16-lane-group reductions (rows of a C fragment live in one 16-lane half)
__device__ inline float rowmax16(float v) {
    for (int m = 1; m < 16; m <<= 1) v = fmaxf(v, __shfl_xor(v, m, 32));
    return v;
}
__device__ inline float rowsum16(float v) {
    for (int m = 1; m < 16; m <<= 1) v += __shfl_xor(v, m, 32);
    return v;
}

// ---------------------------------------------------------------------------
// fp32 -> bf16 convert
// ---------------------------------------------------------------------------
__global__ void k_convert_bf16(const float* __restrict__ in, __bf16* __restrict__ out, int n) {
    int i = blockIdx.x * blockDim.x + threadIdx.x;
    if (i < n) out[i] = f2bf(in[i]);
}

// ---------------------------------------------------------------------------
// Generic WMMA GEMM: C[N][M] = A[N][K] * B[M][K]^T + bias[M]
// mode 0: store fp32 ; mode 1: exact GELU then store bf16
// block = 256 threads (8 waves in 4x2), block tile 256x128, wave tile 64x64,
// K-step 64, double-buffered async global->LDS staging.
// 32 WMMA per K-tile vs 16 fragment loads -> 2x operand reuse.
// ---------------------------------------------------------------------------
#define GK 72   // padded LDS K-stride for 64-wide chunks (144B, multiple of 16B)

__global__ __launch_bounds__(256)
void k_gemm_bf16(const __bf16* __restrict__ A, const __bf16* __restrict__ B,
                 const float* __restrict__ bias, void* __restrict__ Cout,
                 int N, int M, int K, int mode) {
    __shared__ __bf16 As[2][256 * GK];   // 73,728 B
    __shared__ __bf16 Bs[2][128 * GK];   // 36,864 B

    const int tid  = threadIdx.x;
    const int wave = tid >> 5;
    const int lane = tid & 31;
    const int row0 = blockIdx.y * 256;
    const int col0 = blockIdx.x * 128;
    const int wm = wave >> 1;          // 0..3 -> 64-row strip
    const int wn = wave & 1;           // 0..1 -> 64-col strip

    v8f acc[4][4];
    #pragma unroll
    for (int i = 0; i < 4; ++i)
        #pragma unroll
        for (int j = 0; j < 4; ++j)
            #pragma unroll
            for (int e = 0; e < 8; ++e) acc[i][j][e] = 0.0f;

    auto issue = [&](int kt, int buf) {
        // A tile: 256 rows x 64 k = 512 chunks of 64B, 2 per thread
        #pragma unroll
        for (int p = 0; p < 2; ++p) {
            const int c = tid + p * 256;
            const int row = c >> 1, hf = c & 1;
            const __bf16* sa = A + (size_t)(row0 + row) * K + kt + hf * 32;
            __bf16* da = &As[buf][row * GK + hf * 32];
            #pragma unroll
            for (int cc = 0; cc < 4; ++cc) async_copy_b128(da + cc * 8, sa + cc * 8);
        }
        // B tile: 128 rows x 64 k = 256 chunks, 1 per thread
        {
            const int row = tid >> 1, hf = tid & 1;
            const __bf16* sb = B + (size_t)(col0 + row) * K + kt + hf * 32;
            __bf16* db = &Bs[buf][row * GK + hf * 32];
            #pragma unroll
            for (int cc = 0; cc < 4; ++cc) async_copy_b128(db + cc * 8, sb + cc * 8);
        }
    };

    const int ntiles = K >> 6;
    issue(0, 0);
    for (int t = 0; t < ntiles; ++t) {
        wait_async0();
        __syncthreads();
        if (t + 1 < ntiles) issue((t + 1) << 6, (t + 1) & 1);

        const __bf16* Ab = As[t & 1];
        const __bf16* Bb = Bs[t & 1];
        #pragma unroll
        for (int ks = 0; ks < 64; ks += 32) {
            v16bf af[4], bfr[4];
            #pragma unroll
            for (int i = 0; i < 4; ++i)
                af[i] = load_frag_a(Ab, wm * 64 + i * 16, ks, GK);
            #pragma unroll
            for (int j = 0; j < 4; ++j)
                bfr[j] = load_frag_b(Bb, wn * 64 + j * 16, ks, GK);
            #pragma unroll
            for (int i = 0; i < 4; ++i)
                #pragma unroll
                for (int j = 0; j < 4; ++j)
                    acc[i][j] = wmma_bf16(af[i], bfr[j], acc[i][j]);
        }
        __syncthreads();
    }

    const int g = lane >> 4, nl = lane & 15;
    #pragma unroll
    for (int i = 0; i < 4; ++i) {
        #pragma unroll
        for (int j = 0; j < 4; ++j) {
            const int col = col0 + wn * 64 + j * 16 + nl;
            const float bv = bias ? bias[col] : 0.0f;
            #pragma unroll
            for (int e = 0; e < 8; ++e) {
                const int row = row0 + wm * 64 + i * 16 + g * 8 + e;
                float v = acc[i][j][e] + bv;
                if (mode == 0) {
                    ((float*)Cout)[(size_t)row * M + col] = v;
                } else { // exact-erf GELU -> bf16
                    v = 0.5f * v * (1.0f + erff(v * 0.70710678118654752f));
                    ((__bf16*)Cout)[(size_t)row * M + col] = f2bf(v);
                }
            }
        }
    }
}

// ---------------------------------------------------------------------------
// Per-(row, head) L2 normalization: fp32 [8192,1024] -> bf16 [8192,1024]
// ---------------------------------------------------------------------------
__global__ void k_l2norm_bf16(const float* __restrict__ in, __bf16* __restrict__ out) {
    int idx = blockIdx.x * blockDim.x + threadIdx.x;   // row*16 + head
    int row = idx >> 4, head = idx & 15;
    const float* p = in + (size_t)row * D_MODEL + head * DK;
    float ss = 0.0f;
    float v[DK];
    #pragma unroll
    for (int d = 0; d < DK; ++d) { v[d] = p[d]; ss += v[d] * v[d]; }
    float sc = 1.0f / fmaxf(sqrtf(ss), 1e-12f);
    __bf16* q = out + (size_t)row * D_MODEL + head * DK;
    #pragma unroll
    for (int d = 0; d < DK; ++d) q[d] = f2bf(v[d] * sc);
}

// ---------------------------------------------------------------------------
// V transpose: fp32 [b,t,h,d] -> bf16 [b,h,d,t]  (K-contiguous for P*V)
// ---------------------------------------------------------------------------
__global__ void k_vtrans_bf16(const float* __restrict__ V, __bf16* __restrict__ Vt) {
    int idx = blockIdx.x * blockDim.x + threadIdx.x;
    int row = idx >> 4, head = idx & 15;
    int b = row >> 10, t = row & 1023;
    const float* p = V + (size_t)row * D_MODEL + head * DK;
    __bf16* q = Vt + ((size_t)(b * N_HEADS + head) * DK) * SEQ + t;
    #pragma unroll
    for (int d = 0; d < DK; ++d) q[(size_t)d * SEQ] = f2bf(p[d]);
}

// ---------------------------------------------------------------------------
// Gate: gate[b,h] = sigmoid(g[h]) * sigmoid(mean_t(h[b,t,:]·Wg2[h,:]) + bg2[h])
// ---------------------------------------------------------------------------
__global__ __launch_bounds__(256)
void k_gate(const __bf16* __restrict__ hmlp, const float* __restrict__ Wg2,
            const float* __restrict__ bg2, const float* __restrict__ gstat,
            float* __restrict__ gate) {
    __shared__ float red[256];
    int b = blockIdx.x >> 4, hd = blockIdx.x & 15;
    int tid = threadIdx.x;
    const float* w = Wg2 + (size_t)hd * HIDDEN;
    float partial = 0.0f;
    for (int t = tid; t < SEQ; t += 256) {
        const __bf16* hr = hmlp + (size_t)(b * SEQ + t) * HIDDEN;
        float dot = 0.0f;
        for (int c = 0; c < HIDDEN; ++c) dot += bf2f(hr[c]) * w[c];
        partial += dot;
    }
    red[tid] = partial;
    __syncthreads();
    for (int s = 128; s > 0; s >>= 1) {
        if (tid < s) red[tid] += red[tid + s];
        __syncthreads();
    }
    if (tid == 0) {
        float logit = red[0] * (1.0f / (float)SEQ) + bg2[hd];
        float gd = 1.0f / (1.0f + __expf(-logit));
        float gs = 1.0f / (1.0f + __expf(-gstat[hd]));
        gate[b * N_HEADS + hd] = gs * gd;
    }
}

// ---------------------------------------------------------------------------
// Flash attention: one block = (b,h, 64-row q-tile); 128 threads = 4 waves.
// Async double-buffered K/V staging; S = Qn*Kn^T/8 (WMMA); online softmax in
// registers; P staged to LDS bf16; O += P*V (WMMA); gate + 1/l epilogue.
// ---------------------------------------------------------------------------
#define ALDK 72   // 144B stride, multiple of 16B

__global__ __launch_bounds__(128)
void k_attention(const __bf16* __restrict__ Qn, const __bf16* __restrict__ Kn,
                 const __bf16* __restrict__ Vt, const float* __restrict__ gate,
                 __bf16* __restrict__ Hout) {
    __shared__ __bf16 Qs[64 * ALDK];
    __shared__ __bf16 Ks[2][64 * ALDK];
    __shared__ __bf16 Vs[2][64 * ALDK];
    __shared__ __bf16 Ps[64 * ALDK];

    const int tid  = threadIdx.x;
    const int wave = tid >> 5;
    const int lane = tid & 31;
    const int g = lane >> 4, nl = lane & 15;
    const int bh = blockIdx.y;
    const int b = bh >> 4, h = bh & 15;
    const int q0 = blockIdx.x * 64;
    const int r = tid >> 1, half = tid & 1;

    auto issue_kv = [&](int kb, int buf) {
        const __bf16* sk = Kn + (size_t)(b * SEQ + kb * 64 + r) * D_MODEL + h * DK + half * 32;
        __bf16* dk = &Ks[buf][r * ALDK + half * 32];
        const __bf16* sv = Vt + ((size_t)(b * N_HEADS + h) * DK + r) * SEQ + kb * 64 + half * 32;
        __bf16* dv = &Vs[buf][r * ALDK + half * 32];
        #pragma unroll
        for (int c = 0; c < 4; ++c) {
            async_copy_b128(dk + c * 8, sk + c * 8);
            async_copy_b128(dv + c * 8, sv + c * 8);
        }
    };

    // async-stage Q tile (64 rows x 64 d) + first K/V tiles
    {
        const __bf16* sq = Qn + (size_t)(b * SEQ + q0 + r) * D_MODEL + h * DK + half * 32;
        __bf16* dq = Qs + r * ALDK + half * 32;
        #pragma unroll
        for (int c = 0; c < 4; ++c) async_copy_b128(dq + c * 8, sq + c * 8);
    }
    issue_kv(0, 0);

    v8f accO[4];
    float m_old[8], l_row[8];
    #pragma unroll
    for (int j = 0; j < 4; ++j)
        #pragma unroll
        for (int e = 0; e < 8; ++e) accO[j][e] = 0.0f;
    #pragma unroll
    for (int e = 0; e < 8; ++e) { m_old[e] = -3.0e38f; l_row[e] = 0.0f; }

    for (int kb = 0; kb < SEQ / 64; ++kb) {
        wait_async0();
        __syncthreads();
        if (kb + 1 < SEQ / 64) issue_kv(kb + 1, (kb + 1) & 1);
        const __bf16* Kb = Ks[kb & 1];
        const __bf16* Vb = Vs[kb & 1];

        // S = Q*K^T (16 rows per wave, 64 cols, k = 64)
        v8f s4[4];
        #pragma unroll
        for (int j = 0; j < 4; ++j)
            #pragma unroll
            for (int e = 0; e < 8; ++e) s4[j][e] = 0.0f;
        #pragma unroll
        for (int ks = 0; ks < 64; ks += 32) {
            v16bf a = load_frag_a(Qs, wave * 16, ks, ALDK);
            #pragma unroll
            for (int j = 0; j < 4; ++j) {
                v16bf bk = load_frag_b(Kb, j * 16, ks, ALDK);
                s4[j] = wmma_bf16(a, bk, s4[j]);
            }
        }
        #pragma unroll
        for (int j = 0; j < 4; ++j)
            #pragma unroll
            for (int e = 0; e < 8; ++e) s4[j][e] *= 0.125f;   // 1/sqrt(64)

        // online softmax (row = wave*16 + g*8 + e, lives in one 16-lane half)
        float mnew[8], fac[8], psum[8];
        #pragma unroll
        for (int e = 0; e < 8; ++e) {
            float ml = -3.0e38f;
            #pragma unroll
            for (int j = 0; j < 4; ++j) ml = fmaxf(ml, s4[j][e]);
            ml = rowmax16(ml);
            mnew[e] = fmaxf(m_old[e], ml);
            fac[e]  = __expf(m_old[e] - mnew[e]);
            psum[e] = 0.0f;
        }
        #pragma unroll
        for (int j = 0; j < 4; ++j) {
            #pragma unroll
            for (int e = 0; e < 8; ++e) {
                float pv = __expf(s4[j][e] - mnew[e]);
                psum[e] += pv;
                Ps[(size_t)(wave * 16 + g * 8 + e) * ALDK + j * 16 + nl] = f2bf(pv);
            }
        }
        #pragma unroll
        for (int e = 0; e < 8; ++e) {
            psum[e]  = rowsum16(psum[e]);
            l_row[e] = l_row[e] * fac[e] + psum[e];
            m_old[e] = mnew[e];
        }
        #pragma unroll
        for (int j = 0; j < 4; ++j)
            #pragma unroll
            for (int e = 0; e < 8; ++e) accO[j][e] *= fac[e];

        // O += P * V   (P rows written & read by the same wave; LDS in-order)
        #pragma unroll
        for (int ks = 0; ks < 64; ks += 32) {
            v16bf a = load_frag_a(Ps, wave * 16, ks, ALDK);
            #pragma unroll
            for (int j = 0; j < 4; ++j) {
                v16bf bv = load_frag_b(Vb, j * 16, ks, ALDK);
                accO[j] = wmma_bf16(a, bv, accO[j]);
            }
        }
        __syncthreads();
    }

    const float gt = gate[b * N_HEADS + h];
    #pragma unroll
    for (int j = 0; j < 4; ++j) {
        #pragma unroll
        for (int e = 0; e < 8; ++e) {
            const int row = q0 + wave * 16 + g * 8 + e;
            const int col = h * DK + j * 16 + nl;
            float v = accO[j][e] / l_row[e] * gt;
            Hout[(size_t)(b * SEQ + row) * D_MODEL + col] = f2bf(v);
        }
    }
}

// ---------------------------------------------------------------------------
// Host launcher
// ---------------------------------------------------------------------------
extern "C" void kernel_launch(void* const* d_in, const int* in_sizes, int n_in,
                              void* d_out, int out_size, void* d_ws, size_t ws_size,
                              hipStream_t stream) {
    (void)in_sizes; (void)n_in; (void)out_size; (void)ws_size;
    const float* x   = (const float*)d_in[0];
    const float* Wq  = (const float*)d_in[1];
    const float* bq  = (const float*)d_in[2];
    const float* Wk  = (const float*)d_in[3];
    const float* bk  = (const float*)d_in[4];
    const float* Wv  = (const float*)d_in[5];
    const float* bv  = (const float*)d_in[6];
    const float* Wo  = (const float*)d_in[7];
    const float* bo  = (const float*)d_in[8];
    const float* gst = (const float*)d_in[9];
    const float* Wg1 = (const float*)d_in[10];
    const float* bg1 = (const float*)d_in[11];
    const float* Wg2 = (const float*)d_in[12];
    const float* bg2 = (const float*)d_in[13];

    // workspace carve-out (≈203 MB total)
    char* ws = (char*)d_ws;
    size_t off = 0;
    auto alloc = [&](size_t bytes) { void* p = ws + off; off = (off + bytes + 255) & ~(size_t)255; return p; };
    __bf16* xb   = (__bf16*)alloc((size_t)NROWS * D_MODEL * 2);
    __bf16* Wqb  = (__bf16*)alloc((size_t)D_MODEL * D_MODEL * 2);
    __bf16* Wkb  = (__bf16*)alloc((size_t)D_MODEL * D_MODEL * 2);
    __bf16* Wvb  = (__bf16*)alloc((size_t)D_MODEL * D_MODEL * 2);
    __bf16* Wob  = (__bf16*)alloc((size_t)D_MODEL * D_MODEL * 2);
    __bf16* Wg1b = (__bf16*)alloc((size_t)HIDDEN * D_MODEL * 2);
    float*  Qf   = (float*) alloc((size_t)NROWS * D_MODEL * 4);
    float*  Kf   = (float*) alloc((size_t)NROWS * D_MODEL * 4);
    float*  Vf   = (float*) alloc((size_t)NROWS * D_MODEL * 4);
    __bf16* Qnb  = (__bf16*)alloc((size_t)NROWS * D_MODEL * 2);
    __bf16* Knb  = (__bf16*)alloc((size_t)NROWS * D_MODEL * 2);
    __bf16* Vtb  = (__bf16*)alloc((size_t)NROWS * D_MODEL * 2);
    __bf16* hb   = (__bf16*)alloc((size_t)NROWS * HIDDEN * 2);
    __bf16* hob  = (__bf16*)alloc((size_t)NROWS * D_MODEL * 2);
    float*  gate = (float*) alloc((size_t)BATCH * N_HEADS * 4);

    // 1) fp32 -> bf16 conversions
    auto cvt = [&](const float* src, __bf16* dst, int n) {
        k_convert_bf16<<<(n + 255) / 256, 256, 0, stream>>>(src, dst, n);
    };
    cvt(x,   xb,   NROWS * D_MODEL);
    cvt(Wq,  Wqb,  D_MODEL * D_MODEL);
    cvt(Wk,  Wkb,  D_MODEL * D_MODEL);
    cvt(Wv,  Wvb,  D_MODEL * D_MODEL);
    cvt(Wo,  Wob,  D_MODEL * D_MODEL);
    cvt(Wg1, Wg1b, HIDDEN * D_MODEL);

    // 2) projections + gate-hidden GEMMs  (block tile 256x128)
    dim3 blk(256);
    dim3 gProj(D_MODEL / 128, NROWS / 256);
    k_gemm_bf16<<<gProj, blk, 0, stream>>>(xb, Wqb, bq, Qf, NROWS, D_MODEL, D_MODEL, 0);
    k_gemm_bf16<<<gProj, blk, 0, stream>>>(xb, Wkb, bk, Kf, NROWS, D_MODEL, D_MODEL, 0);
    k_gemm_bf16<<<gProj, blk, 0, stream>>>(xb, Wvb, bv, Vf, NROWS, D_MODEL, D_MODEL, 0);
    dim3 gHid(HIDDEN / 128, NROWS / 256);
    k_gemm_bf16<<<gHid, blk, 0, stream>>>(xb, Wg1b, bg1, hb, NROWS, HIDDEN, D_MODEL, 1);

    // 3) normalize Q/K, transpose V
    k_l2norm_bf16<<<(NROWS * N_HEADS) / 256, 256, 0, stream>>>(Qf, Qnb);
    k_l2norm_bf16<<<(NROWS * N_HEADS) / 256, 256, 0, stream>>>(Kf, Knb);
    k_vtrans_bf16<<<(NROWS * N_HEADS) / 256, 256, 0, stream>>>(Vf, Vtb);

    // 4) gates
    k_gate<<<BATCH * N_HEADS, 256, 0, stream>>>(hb, Wg2, bg2, gst, gate);

    // 5) flash attention (gate + softmax-normalize fused)
    dim3 gAtt(SEQ / 64, BATCH * N_HEADS);
    k_attention<<<gAtt, 128, 0, stream>>>(Qnb, Knb, Vtb, gate, hob);

    // 6) output projection -> fp32 d_out
    k_gemm_bf16<<<gProj, blk, 0, stream>>>(hob, Wob, bo, (float*)d_out, NROWS, D_MODEL, D_MODEL, 0);
}
